// CustomCLIP_3547642986665
// MI455X (gfx1250) — compile-verified, matrix operand order checked by reading, not run on hardware
//
#include <hip/hip_runtime.h>
#include <hip/hip_bf16.h>

// ---------------------------------------------------------------------------
// CDNA5 (gfx1250) bf16 WMMA implementation of the routed-adapter CLIP head.
//   h = relu(X @ W1[d]); a = relu(h @ W2[d]); routed per-sample gather
//   f = 0.2*a + 0.8*x;  f /= ||f||;  logits = exp(ls) * f @ txtN^T
// ---------------------------------------------------------------------------

typedef __attribute__((ext_vector_type(16))) __bf16 v16bf;
typedef __attribute__((ext_vector_type(8)))  __bf16 bf16x8;
typedef __attribute__((ext_vector_type(8)))  float  v8f;

#define WMMA_BF16(A, B, C) \
  __builtin_amdgcn_wmma_f32_16x16x32_bf16(false, (A), false, (B), (short)0, (C), false, false)

// K index inside a 32-wide K-slab for A/B 16-bit WMMA layout:
//   element e (0..15) of a lane:  e<8 -> klo+e ;  e>=8 -> 16+klo+(e-8)
// with klo = 0 for lanes 0..15, 8 for lanes 16..31.
__device__ __forceinline__ int kmap_e(int klo, int e) {
  return (e < 8) ? (klo + e) : (16 + klo + (e - 8));
}

__device__ __forceinline__ v16bf combine8(bf16x8 lo, bf16x8 hi) {
  v16bf r;
#pragma unroll
  for (int e = 0; e < 8; ++e) { r[e] = lo[e]; r[8 + e] = hi[e]; }
  return r;
}

// Build an A fragment from a row of f32 data (two contiguous 8-float runs).
__device__ __forceinline__ v16bf cvt_a_f32(const float* p) {
  float4 a0 = *(const float4*)(p + 0);
  float4 a1 = *(const float4*)(p + 4);
  float4 a2 = *(const float4*)(p + 16);
  float4 a3 = *(const float4*)(p + 20);
  v16bf r;
  r[0]  = (__bf16)a0.x; r[1]  = (__bf16)a0.y; r[2]  = (__bf16)a0.z; r[3]  = (__bf16)a0.w;
  r[4]  = (__bf16)a1.x; r[5]  = (__bf16)a1.y; r[6]  = (__bf16)a1.z; r[7]  = (__bf16)a1.w;
  r[8]  = (__bf16)a2.x; r[9]  = (__bf16)a2.y; r[10] = (__bf16)a2.z; r[11] = (__bf16)a2.w;
  r[12] = (__bf16)a3.x; r[13] = (__bf16)a3.y; r[14] = (__bf16)a3.z; r[15] = (__bf16)a3.w;
  return r;
}

// ---------------------------------------------------------------------------
// 1) rnorm[n] = exp(logit_scale) / ||txt[n]||   (one wave per text row)
// ---------------------------------------------------------------------------
__global__ void txt_rnorm_kernel(const float* __restrict__ txt,
                                 const float* __restrict__ logit_scale,
                                 float* __restrict__ rnorm, int n_text, int C) {
  int wid  = (blockIdx.x * blockDim.x + threadIdx.x) >> 5;
  int lane = threadIdx.x & 31;
  if (wid >= n_text) return;
  const float* p = txt + (size_t)wid * C;
  float s = 0.f;
  for (int i = lane; i < C; i += 32) { float v = p[i]; s += v * v; }
#pragma unroll
  for (int m = 16; m; m >>= 1) s += __shfl_xor(s, m, 32);
  if (lane == 0) rnorm[wid] = expf(logit_scale[0]) * rsqrtf(s);
}

// ---------------------------------------------------------------------------
// 2) Pack normalized text^T (K=C, N=n_text padded to 16) into WMMA-B tiles:
//    tile t = nt*nk + kt, 512 bf16 per tile, lane-major (lane*16 contiguous).
// ---------------------------------------------------------------------------
__global__ void pack_txt_kernel(const float* __restrict__ txt,
                                const float* __restrict__ rnorm,
                                __bf16* __restrict__ dst, int n_text, int C, int nk) {
  int t    = blockIdx.x;
  int lane = threadIdx.x;
  int nt   = t / nk, kt = t % nk;
  int n    = nt * 16 + (lane & 15);
  int klo  = (lane & 16) ? 8 : 0;
  float scale = (n < n_text) ? rnorm[n] : 0.f;
  const float* src = txt + (size_t)((n < n_text) ? n : 0) * C + kt * 32;
  __bf16* o = dst + (size_t)t * 512 + lane * 16;
#pragma unroll
  for (int e = 0; e < 16; ++e) o[e] = (__bf16)(src[kmap_e(klo, e)] * scale);
}

// ---------------------------------------------------------------------------
// 3) Pack W[d] (K x N row-major, D stacked) into WMMA-B tiles,
//    tile order t = d*nn*nk + nt*nk + kt.
// ---------------------------------------------------------------------------
__global__ void pack_w_kernel(const float* __restrict__ W, __bf16* __restrict__ dst,
                              int K, int N, int nk, int nn) {
  int t    = blockIdx.x;
  int lane = threadIdx.x;
  int d    = t / (nn * nk);
  int r    = t % (nn * nk);
  int nt   = r / nk, kt = r % nk;
  int n    = nt * 16 + (lane & 15);
  int klo  = (lane & 16) ? 8 : 0;
  const float* src = W + (size_t)d * K * N + (size_t)(kt * 32) * N + n;
  __bf16* o = dst + (size_t)t * 512 + lane * 16;
#pragma unroll
  for (int e = 0; e < 16; ++e) o[e] = (__bf16)src[(size_t)kmap_e(klo, e) * N];
}

// ---------------------------------------------------------------------------
// 4) Adapter: one wave per 16-sample tile.
//    GEMM1 (16x512 @ 512x128) -> relu -> LDS, GEMM2 (16x128 @ 128x512) in
//    64-col chunks -> relu -> routed select + blend -> unnormalized feats f32.
// ---------------------------------------------------------------------------
__global__ void __launch_bounds__(32)
adapter_kernel(const float* __restrict__ X, const int* __restrict__ lab,
               const __bf16* __restrict__ W1p, const __bf16* __restrict__ W2p,
               float* __restrict__ featsF) {
  __shared__ float  sX[16 * 512];   // X tile, f32
  __shared__ __bf16 sH[16 * 128];   // hidden tile, bf16

  const int lane = threadIdx.x;
  const int m0   = blockIdx.x * 16;

  // cooperative coalesced X-tile load
  const float4* xin = (const float4*)(X + (size_t)m0 * 512);
  float4* xs = (float4*)sX;
#pragma unroll
  for (int i = 0; i < 64; ++i) xs[i * 32 + lane] = xin[i * 32 + lane];

  int labv[16];
  int mask = 0;
#pragma unroll
  for (int r = 0; r < 16; ++r) { labv[r] = lab[m0 + r]; mask |= 1 << labv[r]; }
  __syncthreads();

  const int row = lane & 15;
  const int klo = (lane & 16) ? 8 : 0;

  for (int d = 0; d < 3; ++d) {
    if (!((mask >> d) & 1)) continue;   // wave-uniform branch

    // ---- GEMM1: H = relu(Xtile @ W1[d]),  K=512 (16 slabs), N=128 (8 tiles)
    v8f acc[8] = {};
    for (int kt = 0; kt < 16; ++kt) {
      v16bf A = cvt_a_f32(sX + row * 512 + kt * 32 + klo);
#pragma unroll
      for (int nt = 0; nt < 8; ++nt) {
        v16bf B = *(const v16bf*)(W1p + (size_t)(((d * 8 + nt) * 16 + kt) * 512 + lane * 16));
        acc[nt] = WMMA_BF16(A, B, acc[nt]);
      }
    }
    __syncthreads();   // WAR vs previous domain's sH reads
#pragma unroll
    for (int nt = 0; nt < 8; ++nt)
#pragma unroll
      for (int i = 0; i < 8; ++i) {
        int r2 = (lane < 16) ? i : (i + 8);
        float v = acc[nt][i];
        sH[r2 * 128 + nt * 16 + row] = (__bf16)(v > 0.f ? v : 0.f);
      }
    __syncthreads();

    // ---- GEMM2: A = relu(H @ W2[d]),  K=128 (4 slabs), N=512 in 64-col chunks
    for (int nc = 0; nc < 8; ++nc) {
      v8f acc2[4] = {};
#pragma unroll
      for (int kt = 0; kt < 4; ++kt) {
        const __bf16* q = sH + row * 128 + kt * 32 + klo;
        v16bf A2 = combine8(*(const bf16x8*)q, *(const bf16x8*)(q + 16));
#pragma unroll
        for (int j = 0; j < 4; ++j) {
          v16bf B = *(const v16bf*)(W2p + (size_t)(((d * 32 + nc * 4 + j) * 4 + kt) * 512 + lane * 16));
          acc2[j] = WMMA_BF16(A2, B, acc2[j]);
        }
      }
      // routed select + blend (lane-predicated stores, outside WMMA)
#pragma unroll
      for (int j = 0; j < 4; ++j)
#pragma unroll
        for (int i = 0; i < 8; ++i) {
          int r2 = (lane < 16) ? i : (i + 8);
          if (labv[r2] == d) {
            int col = (nc * 4 + j) * 16 + row;
            float a = acc2[j][i];
            a = a > 0.f ? a : 0.f;
            featsF[(size_t)(m0 + r2) * 512 + col] = 0.2f * a + 0.8f * sX[r2 * 512 + col];
          }
        }
    }
  }
}

// ---------------------------------------------------------------------------
// 5) Row-normalize feats, emit bf16 (one wave per row)
// ---------------------------------------------------------------------------
__global__ void norm_feats_kernel(const float* __restrict__ featsF,
                                  __bf16* __restrict__ featsB) {
  int row  = (blockIdx.x * blockDim.x + threadIdx.x) >> 5;
  int lane = threadIdx.x & 31;
  const float* p = featsF + (size_t)row * 512;
  float v[16];
  float s = 0.f;
#pragma unroll
  for (int i = 0; i < 16; ++i) { v[i] = p[lane + i * 32]; s += v[i] * v[i]; }
#pragma unroll
  for (int m = 16; m; m >>= 1) s += __shfl_xor(s, m, 32);
  float r = rsqrtf(s);
  __bf16* o = featsB + (size_t)row * 512;
#pragma unroll
  for (int i = 0; i < 16; ++i) o[lane + i * 32] = (__bf16)(v[i] * r);
}

// ---------------------------------------------------------------------------
// 6) logits = featsB @ txtp  (scale already folded into txtp)
//    one wave -> 16 rows x (J*16) cols, K = 512 (16 slabs), fully unrolled.
//    CHECK: predicate column stores against n_text (tail tiles only).
// ---------------------------------------------------------------------------
template <int J, bool CHECK>
__global__ void __launch_bounds__(32)
logits_kernel(const __bf16* __restrict__ featsB, const __bf16* __restrict__ txtp,
              float* __restrict__ out, int n_text, int ntBase0) {
  const int lane   = threadIdx.x;
  const int m0     = blockIdx.x * 16;
  const int ntBase = ntBase0 + blockIdx.y * J;
  const int row    = lane & 15;
  const int klo    = (lane & 16) ? 8 : 0;

  v8f acc[J] = {};
  const __bf16* arow = featsB + (size_t)(m0 + row) * 512;
  const __bf16* bbase = txtp + (size_t)ntBase * 16 * 512 + lane * 16;
  for (int kt = 0; kt < 16; ++kt) {
    const __bf16* p = arow + kt * 32 + klo;
    v16bf A = combine8(*(const bf16x8*)p, *(const bf16x8*)(p + 16));
    const __bf16* bk = bbase + (size_t)kt * 512;
#pragma unroll
    for (int j = 0; j < J; ++j) {
      v16bf B = *(const v16bf*)(bk + (size_t)j * 16 * 512);
      acc[j] = WMMA_BF16(A, B, acc[j]);
    }
  }
#pragma unroll
  for (int j = 0; j < J; ++j) {
    int col = (ntBase + j) * 16 + row;
#pragma unroll
    for (int i = 0; i < 8; ++i) {
      int r2 = (lane < 16) ? i : (i + 8);
      if (!CHECK || col < n_text)
        out[(size_t)(m0 + r2) * n_text + col] = acc[j][i];
    }
  }
}

// ---------------------------------------------------------------------------
extern "C" void kernel_launch(void* const* d_in, const int* in_sizes, int n_in,
                              void* d_out, int out_size, void* d_ws, size_t ws_size,
                              hipStream_t stream) {
  const float* X   = (const float*)d_in[0];   // [16384, 512]
  const int*   lab = (const int*)  d_in[1];   // [16384]
  const float* W1  = (const float*)d_in[2];   // [3, 512, 128]
  const float* W2  = (const float*)d_in[3];   // [3, 128, 512]
  const float* txt = (const float*)d_in[4];   // [1380, 512]
  const float* ls  = (const float*)d_in[5];   // [1]
  float* out = (float*)d_out;                 // [16384, 1380]

  const int B = 16384, C = 512, Hd = 128, NT = 1380;
  const int nkT = C / 32;            // 16 K-slabs for text GEMM
  const int nnT = (NT + 15) / 16;    // 87 N-tiles (padded to 1392)
  (void)Hd; (void)in_sizes; (void)n_in; (void)out_size; (void)ws_size;

  char* ws = (char*)d_ws;
  size_t off = 0;
  auto alloc = [&](size_t bytes) -> char* {
    char* p = ws + off;
    off = (off + bytes + 255) & ~(size_t)255;
    return p;
  };
  __bf16* txtp   = (__bf16*)alloc((size_t)nnT * nkT * 512 * 2);   // ~1.4 MB
  __bf16* w1p    = (__bf16*)alloc((size_t)3 * 8 * 16 * 512 * 2);  // 384 KB
  __bf16* w2p    = (__bf16*)alloc((size_t)3 * 32 * 4 * 512 * 2);  // 384 KB
  float*  rnorm  = (float*) alloc((size_t)NT * 4);
  float*  featsF = (float*) alloc((size_t)B * C * 4);             // 32 MB
  __bf16* featsB = (__bf16*)alloc((size_t)B * C * 2);             // 16 MB

  txt_rnorm_kernel<<<(NT * 32 + 255) / 256, 256, 0, stream>>>(txt, ls, rnorm, NT, C);
  pack_txt_kernel<<<nnT * nkT, 32, 0, stream>>>(txt, rnorm, txtp, NT, C, nkT);
  pack_w_kernel<<<3 * 8 * 16, 32, 0, stream>>>(W1, w1p, /*K=*/C, /*N=*/128, /*nk=*/16, /*nn=*/8);
  pack_w_kernel<<<3 * 32 * 4, 32, 0, stream>>>(W2, w2p, /*K=*/128, /*N=*/C, /*nk=*/4, /*nn=*/32);
  adapter_kernel<<<B / 16, 32, 0, stream>>>(X, lab, w1p, w2p, featsF);
  norm_feats_kernel<<<B / 8, 256, 0, stream>>>(featsF, featsB);

  // logits: fast path covers N-tiles [0, 80) with 16x128 waves (no bounds
  // checks needed: max col 1279 < 1380); tail covers tiles [80, 87).
  logits_kernel<8, false><<<dim3(B / 16, 10), 32, 0, stream>>>(featsB, txtp, out, NT, 0);
  logits_kernel<7, true ><<<dim3(B / 16, 1),  32, 0, stream>>>(featsB, txtp, out, NT, 80);
}
